// SwinThyroidClassifier_63050119905523
// MI455X (gfx1250) — matrix operand order repaired and checked by reading, not compiled, not run on hardware
//
#include <hip/hip_runtime.h>
#include <cstdint>
#include <cstddef>

// ---------------------------------------------------------------------------
// Swin (2 blocks, shifted windows) + classifier head for MI455X (gfx1250).
// Big GEMMs run on v_wmma_f32_16x16x32_bf16 (wave32 WMMA), 128x128 block tile,
// LDS double buffering (1 barrier per K-step), 2-deep global pipeline,
// pre-transposed bf16 weights. Residual stream, layernorm, softmax stay f32.
// ---------------------------------------------------------------------------

#define Bsz   64
#define Himg  14
#define Wimg  14
#define Cdim  1024
#define WS    7
#define SHIFT 3
#define NHEAD 32
#define HDIM  32
#define TOK   (Himg * Wimg)          // 196
#define MROWS (Bsz * TOK)            // 12544
#define NWIN  (Bsz * 4)              // 256 windows
#define WTOK  (WS * WS)              // 49
#define SCALE 0.17677669529663687f   // 32^-0.5

typedef __attribute__((ext_vector_type(16))) __bf16 v16bf;
typedef __attribute__((ext_vector_type(8)))  float  v8f;

union FragBF { v16bf v; unsigned int u[8]; };

__device__ __forceinline__ float gelu_exact(float x) {
  return 0.5f * x * (1.0f + erff(x * 0.7071067811865475f));
}

// ---------------------------------------------------------------------------
// Weight prep: f32 [K,N] -> bf16 transposed [N,K] (done once; makes the GEMM
// B-tile loader a straight global_load_b128 -> ds_store_b128 path).
// ---------------------------------------------------------------------------
__global__ __launch_bounds__(256) void cvt_t_bf16_kernel(const float* __restrict__ in,
                                                         __bf16* __restrict__ out,
                                                         int K, int N) {
  long i = (long)blockIdx.x * 256 + threadIdx.x;
  if (i < (long)K * N) {
    int n = (int)(i / K), k = (int)(i % K);
    out[i] = (__bf16)in[(size_t)k * N + n];
  }
}

// ---------------------------------------------------------------------------
// LayerNorm + (shifted) window partition: x [B,196,C] f32 -> hbf [12544,C] bf16
// ---------------------------------------------------------------------------
__global__ __launch_bounds__(256) void ln_part_kernel(const float* __restrict__ x,
                                                      const float* __restrict__ g,
                                                      const float* __restrict__ b,
                                                      __bf16* __restrict__ out,
                                                      int do_shift) {
  __shared__ float s1[256], s2[256];
  int r = blockIdx.x;                 // windowed row 0..12543
  int w = r / WTOK, t = r % WTOK;
  int batch = w >> 2, wi = w & 3;
  int h  = (wi >> 1) * WS + t / WS;
  int wc = (wi & 1)  * WS + t % WS;
  if (do_shift) { h = (h + SHIFT) % Himg; wc = (wc + SHIFT) % Wimg; }
  const float* row = x + ((size_t)batch * TOK + h * Wimg + wc) * Cdim;

  int tid = threadIdx.x;
  float sum = 0.f, sq = 0.f;
  for (int c = tid; c < Cdim; c += 256) { float v = row[c]; sum += v; sq += v * v; }
  s1[tid] = sum; s2[tid] = sq; __syncthreads();
  for (int off = 128; off > 0; off >>= 1) {
    if (tid < off) { s1[tid] += s1[tid + off]; s2[tid] += s2[tid + off]; }
    __syncthreads();
  }
  float mu  = s1[0] * (1.0f / Cdim);
  float var = s2[0] * (1.0f / Cdim) - mu * mu;
  float rstd = rsqrtf(var + 1e-5f);
  __bf16* orow = out + (size_t)r * Cdim;
  for (int c = tid; c < Cdim; c += 256)
    orow[c] = (__bf16)((row[c] - mu) * rstd * g[c] + b[c]);
}

// Plain layernorm (original token order) -> bf16
__global__ __launch_bounds__(256) void ln_kernel(const float* __restrict__ x,
                                                 const float* __restrict__ g,
                                                 const float* __restrict__ b,
                                                 __bf16* __restrict__ out) {
  __shared__ float s1[256], s2[256];
  int r = blockIdx.x;
  const float* row = x + (size_t)r * Cdim;
  int tid = threadIdx.x;
  float sum = 0.f, sq = 0.f;
  for (int c = tid; c < Cdim; c += 256) { float v = row[c]; sum += v; sq += v * v; }
  s1[tid] = sum; s2[tid] = sq; __syncthreads();
  for (int off = 128; off > 0; off >>= 1) {
    if (tid < off) { s1[tid] += s1[tid + off]; s2[tid] += s2[tid + off]; }
    __syncthreads();
  }
  float mu  = s1[0] * (1.0f / Cdim);
  float var = s2[0] * (1.0f / Cdim) - mu * mu;
  float rstd = rsqrtf(var + 1e-5f);
  __bf16* orow = out + (size_t)r * Cdim;
  for (int c = tid; c < Cdim; c += 256)
    orow[c] = (__bf16)((row[c] - mu) * rstd * g[c] + b[c]);
}

// ---------------------------------------------------------------------------
// bf16 WMMA GEMM: out[M,N] = A[M,K] @ W[K,N] + bias (+gelu), W given
// pre-transposed as WT[N,K]. Block tile 128x128, BK=32, 256 threads (8 waves
// in a 4x2 grid); each wave owns a 32x64 tile = 8 accumulators -> 8 WMMAs per
// K-step. LDS double-buffered: store tile k+1 to buf^1 while computing buf,
// single barrier per K-step; regs hold tile k+2 (2-deep global pipeline).
// Requires M%128==0, N%128==0, K%32==0 (true at all call sites).
// ---------------------------------------------------------------------------
#define BM 128
#define BN 128
#define BK 32
#define LDP 40   // padded LDS row pitch in bf16 (32 data + 8 pad, 80B)

__global__ __launch_bounds__(256) void gemm_bf16_kernel(const __bf16* __restrict__ A,
                                                        const __bf16* __restrict__ WT,
                                                        const float* __restrict__ bias,
                                                        void* __restrict__ out,
                                                        int M, int N, int K,
                                                        int out_bf16, int act_gelu) {
  __shared__ __bf16 As[2][BM][LDP];  // A tile rows (m, k), double buffered
  __shared__ __bf16 Bs[2][BN][LDP];  // W tile rows (n, k)  -- WT is [N,K]

  int tid  = threadIdx.x;
  int wave = tid >> 5;
  int lane = tid & 31;
  int wm   = wave & 3;               // 0..3  -> 32-row slab
  int wn   = wave >> 2;              // 0..1  -> 64-col slab
  int mBase = blockIdx.y * BM;
  int nBase = blockIdx.x * BN;

  v8f acc[2][4] = {};

  int lRow = tid >> 1;               // 0..127
  int lCol = (tid & 1) * 16;         // 0 or 16

  const __bf16* aPtr = A  + (size_t)(mBase + lRow) * K + lCol;
  const __bf16* bPtr = WT + (size_t)(nBase + lRow) * K + lCol;

  int KT = K / BK;
  int khalf = lane >> 4;
  int l15   = lane & 15;

  // prologue: tile 0 -> LDS[0]; tile 1 -> regs
  uint4 aReg0 = ((const uint4*)aPtr)[0], aReg1 = ((const uint4*)aPtr)[1];
  uint4 bReg0 = ((const uint4*)bPtr)[0], bReg1 = ((const uint4*)bPtr)[1];
  ((uint4*)&As[0][lRow][lCol])[0] = aReg0;
  ((uint4*)&As[0][lRow][lCol])[1] = aReg1;
  ((uint4*)&Bs[0][lRow][lCol])[0] = bReg0;
  ((uint4*)&Bs[0][lRow][lCol])[1] = bReg1;
  if (KT > 1) {
    const __bf16* ap = aPtr + BK;
    const __bf16* bp = bPtr + BK;
    aReg0 = ((const uint4*)ap)[0]; aReg1 = ((const uint4*)ap)[1];
    bReg0 = ((const uint4*)bp)[0]; bReg1 = ((const uint4*)bp)[1];
  }
  __syncthreads();

  for (int kt = 0; kt < KT; ++kt) {
    int cur = kt & 1, nxt = cur ^ 1;
    if (kt + 1 < KT) {
      // stage tile kt+1 into the alternate buffer (read last at kt-1; safe)
      ((uint4*)&As[nxt][lRow][lCol])[0] = aReg0;
      ((uint4*)&As[nxt][lRow][lCol])[1] = aReg1;
      ((uint4*)&Bs[nxt][lRow][lCol])[0] = bReg0;
      ((uint4*)&Bs[nxt][lRow][lCol])[1] = bReg1;
      if (kt + 2 < KT) {
        // issue tile kt+2 global loads; latency hidden under the WMMAs
        const __bf16* ap = aPtr + (kt + 2) * BK;
        const __bf16* bp = bPtr + (kt + 2) * BK;
        aReg0 = ((const uint4*)ap)[0]; aReg1 = ((const uint4*)ap)[1];
        bReg0 = ((const uint4*)bp)[0]; bReg1 = ((const uint4*)bp)[1];
        if (kt + 3 < KT) {
          __builtin_prefetch(aPtr + (kt + 3) * BK, 0, 1);
          __builtin_prefetch(bPtr + (kt + 3) * BK, 0, 1);
        }
      }
    }

    // A fragments (ISA 16-bit A 16x32 layout: lanes 0-15 K{0..7,16..23},
    // lanes 16-31 K{8..15,24..31}); pairs contiguous -> ds_load_b128 x2.
    FragBF a[2];
#pragma unroll
    for (int mt = 0; mt < 2; ++mt) {
      const unsigned int* row = (const unsigned int*)As[cur][(wm << 5) + (mt << 4) + l15];
#pragma unroll
      for (int v = 0; v < 8; ++v)
        a[mt].u[v] = row[(v < 4) ? (v + 4 * khalf) : (v + 4 + 4 * khalf)];
    }
    // B fragments (32x16: lanes 0-15 K{0..7,16..23}, lanes 16-31 K{8..15,24..31})
    FragBF b[4];
#pragma unroll
    for (int nt = 0; nt < 4; ++nt) {
      const unsigned int* row = (const unsigned int*)Bs[cur][(wn << 6) + (nt << 4) + l15];
#pragma unroll
      for (int v = 0; v < 8; ++v) b[nt].u[v] = row[(v < 4) ? (v + 4 * khalf) : (v + 4 + 4 * khalf)];
    }
#pragma unroll
    for (int mt = 0; mt < 2; ++mt)
#pragma unroll
      for (int nt = 0; nt < 4; ++nt)
        acc[mt][nt] = __builtin_amdgcn_wmma_f32_16x16x32_bf16(
            false, a[mt].v, false, b[nt].v, (short)0, acc[mt][nt], false, false);

    // single barrier: publishes buf^1 stores AND retires buf reads
    __syncthreads();
  }

  // Epilogue: D layout — col = lane%16; VGPR v -> row v (lanes 0-15) / v+8.
#pragma unroll
  for (int nt = 0; nt < 4; ++nt) {
    int col = nBase + (wn << 6) + (nt << 4) + l15;
    float bv = bias[col];
#pragma unroll
    for (int mt = 0; mt < 2; ++mt) {
#pragma unroll
      for (int v = 0; v < 8; ++v) {
        int row = mBase + (wm << 5) + (mt << 4) + v + (khalf << 3);
        float val = acc[mt][nt][v] + bv;
        if (act_gelu) val = gelu_exact(val);
        if (out_bf16) ((__bf16*)out)[(size_t)row * N + col] = (__bf16)val;
        else          ((float*)out)[(size_t)row * N + col]  = val;
      }
    }
  }
}

// ---------------------------------------------------------------------------
// Windowed attention: one block per (window, head). qkv [12544,3072] bf16.
// S = (q*scale)k^T + relposbias (+shift mask); softmax; O = S v -> out bf16
// [12544, 1024] in (token, head*32+d) order.
// ---------------------------------------------------------------------------
__global__ __launch_bounds__(64) void attn_kernel(const __bf16* __restrict__ qkv,
                                                  const float* __restrict__ rpb,
                                                  __bf16* __restrict__ out,
                                                  int do_shift) {
  __shared__ float Qs[WTOK][HDIM];
  __shared__ float Ks[WTOK][HDIM];
  __shared__ float Vs[WTOK][HDIM];
  __shared__ float S[WTOK][WTOK + 1];

  int blk  = blockIdx.x;
  int w    = blk >> 5;          // window 0..255
  int head = blk & 31;
  int tid  = threadIdx.x;

  for (int i = tid; i < WTOK * HDIM; i += 64) {
    int t = i / HDIM, d = i % HDIM;
    size_t row = (size_t)(w * WTOK + t) * (3 * Cdim) + head * HDIM + d;
    Qs[t][d] = (float)qkv[row] * SCALE;
    Ks[t][d] = (float)qkv[row + Cdim];
    Vs[t][d] = (float)qkv[row + 2 * Cdim];
  }
  __syncthreads();

  int q = tid;
  if (q < WTOK) {
    int wi = w & 3;
    int qh = (wi >> 1) * WS + q / WS;
    int qw = (wi & 1)  * WS + q % WS;
    int qlab = ((qh < WS) ? 0 : (qh < Himg - SHIFT) ? 1 : 2) * 3 +
               ((qw < WS) ? 0 : (qw < Wimg - SHIFT) ? 1 : 2);
    float mx = -1e30f;
    for (int k = 0; k < WTOK; ++k) {
      float dot = 0.f;
#pragma unroll 8
      for (int d = 0; d < HDIM; ++d) dot += Qs[q][d] * Ks[k][d];
      int dr = q / WS - k / WS + (WS - 1);
      int dc = q % WS - k % WS + (WS - 1);
      dot += rpb[(dr * (2 * WS - 1) + dc) * NHEAD + head];
      if (do_shift) {
        int kh = (wi >> 1) * WS + k / WS;
        int kw = (wi & 1)  * WS + k % WS;
        int klab = ((kh < WS) ? 0 : (kh < Himg - SHIFT) ? 1 : 2) * 3 +
                   ((kw < WS) ? 0 : (kw < Wimg - SHIFT) ? 1 : 2);
        if (klab != qlab) dot -= 100.0f;
      }
      S[q][k] = dot;
      mx = fmaxf(mx, dot);
    }
    float sum = 0.f;
    for (int k = 0; k < WTOK; ++k) { float e = expf(S[q][k] - mx); S[q][k] = e; sum += e; }
    float inv = 1.0f / sum;
    __bf16* orow = out + (size_t)(w * WTOK + q) * Cdim + head * HDIM;
    for (int d = 0; d < HDIM; ++d) {
      float o = 0.f;
#pragma unroll 7
      for (int k = 0; k < WTOK; ++k) o += S[q][k] * Vs[k][d];
      orow[d] = (__bf16)(o * inv);
    }
  }
}

// ---------------------------------------------------------------------------
// window reverse (+roll back) + residual: xres[map(r)] += proj[r]
// ---------------------------------------------------------------------------
__global__ __launch_bounds__(256) void rev_res_kernel(float* __restrict__ xres,
                                                      const float* __restrict__ proj,
                                                      int do_shift) {
  int r = blockIdx.x;
  int w = r / WTOK, t = r % WTOK;
  int batch = w >> 2, wi = w & 3;
  int h  = (wi >> 1) * WS + t / WS;
  int wc = (wi & 1)  * WS + t % WS;
  if (do_shift) { h = (h + SHIFT) % Himg; wc = (wc + SHIFT) % Wimg; }
  size_t dst = ((size_t)batch * TOK + h * Wimg + wc) * Cdim;
  size_t src = (size_t)r * Cdim;
  for (int c = threadIdx.x; c < Cdim; c += 256)
    xres[dst + c] += proj[src + c];
}

__global__ __launch_bounds__(256) void res_add_kernel(float* __restrict__ xres,
                                                      const float* __restrict__ t, int n) {
  int i = blockIdx.x * 256 + threadIdx.x;
  if (i < n) xres[i] += t[i];
}

// ---------------------------------------------------------------------------
// Global average pool over 196 tokens -> pooled [64, 1024]
// ---------------------------------------------------------------------------
__global__ __launch_bounds__(256) void pool_kernel(const float* __restrict__ x,
                                                   float* __restrict__ pooled) {
  int b = blockIdx.x;
  for (int c = threadIdx.x; c < Cdim; c += 256) {
    float s = 0.f;
    const float* p = x + (size_t)b * TOK * Cdim + c;
    for (int t = 0; t < TOK; ++t) s += p[(size_t)t * Cdim];
    pooled[(size_t)b * Cdim + c] = s * (1.0f / TOK);
  }
}

// ---------------------------------------------------------------------------
// Head: LN -> gelu(1024x512) -> gelu(512x128) -> 128x2. One block per batch.
// ---------------------------------------------------------------------------
__global__ __launch_bounds__(256) void head_kernel(const float* __restrict__ pooled,
                                                   const float* __restrict__ g,
                                                   const float* __restrict__ b,
                                                   const float* __restrict__ w1,
                                                   const float* __restrict__ b1,
                                                   const float* __restrict__ w2,
                                                   const float* __restrict__ b2,
                                                   const float* __restrict__ w3,
                                                   const float* __restrict__ b3,
                                                   float* __restrict__ out) {
  __shared__ float buf[1024];
  __shared__ float h1[512];
  __shared__ float h2[128];
  __shared__ float s1[256], s2[256];
  int bb = blockIdx.x, tid = threadIdx.x;
  const float* row = pooled + (size_t)bb * Cdim;
  float sum = 0.f, sq = 0.f;
  for (int c = tid; c < Cdim; c += 256) { float v = row[c]; sum += v; sq += v * v; }
  s1[tid] = sum; s2[tid] = sq; __syncthreads();
  for (int off = 128; off > 0; off >>= 1) {
    if (tid < off) { s1[tid] += s1[tid + off]; s2[tid] += s2[tid + off]; }
    __syncthreads();
  }
  float mu  = s1[0] * (1.0f / Cdim);
  float var = s2[0] * (1.0f / Cdim) - mu * mu;
  float rstd = rsqrtf(var + 1e-5f);
  for (int c = tid; c < Cdim; c += 256) buf[c] = (row[c] - mu) * rstd * g[c] + b[c];
  __syncthreads();
  for (int j = tid; j < 512; j += 256) {
    float a = b1[j];
    for (int c = 0; c < 1024; ++c) a += buf[c] * w1[c * 512 + j];
    h1[j] = gelu_exact(a);
  }
  __syncthreads();
  if (tid < 128) {
    float a = b2[tid];
    for (int c = 0; c < 512; ++c) a += h1[c] * w2[c * 128 + tid];
    h2[tid] = gelu_exact(a);
  }
  __syncthreads();
  if (tid < 2) {
    float a = b3[tid];
    for (int c = 0; c < 128; ++c) a += h2[c] * w3[c * 2 + tid];
    out[bb * 2 + tid] = a;
  }
}

// ---------------------------------------------------------------------------
// Host orchestration
// ---------------------------------------------------------------------------
static inline void launch_gemm(const __bf16* A, const __bf16* WT, const float* bias,
                               void* out, int M, int N, int K, int out_bf16,
                               int act_gelu, hipStream_t stream) {
  dim3 grid(N / BN, M / BM);
  gemm_bf16_kernel<<<grid, 256, 0, stream>>>(A, WT, bias, out, M, N, K, out_bf16, act_gelu);
}

extern "C" void kernel_launch(void* const* d_in, const int* in_sizes, int n_in,
                              void* d_out, int out_size, void* d_ws, size_t ws_size,
                              hipStream_t stream) {
  (void)in_sizes; (void)n_in; (void)out_size; (void)ws_size;

  char* ws = (char*)d_ws;
  size_t off = 0;
  auto alloc = [&](size_t bytes) -> void* {
    void* p = ws + off;
    off += (bytes + 255) & ~(size_t)255;
    return p;
  };

  float*  xres    = (float*) alloc((size_t)MROWS * Cdim * 4);       // residual stream
  __bf16* hbf     = (__bf16*)alloc((size_t)MROWS * Cdim * 2);       // LN output (bf16)
  __bf16* qkvbuf  = (__bf16*)alloc((size_t)MROWS * 3 * Cdim * 2);   // QKV (bf16)
  __bf16* attnout = (__bf16*)alloc((size_t)MROWS * Cdim * 2);       // attention out
  float*  tmpf    = (float*) alloc((size_t)MROWS * Cdim * 4);       // proj / fc2 out
  __bf16* mlpbuf  = (__bf16*)alloc((size_t)MROWS * 4 * Cdim * 2);   // fc1 out (bf16)
  __bf16* wqkv    = (__bf16*)alloc((size_t)Cdim * 3 * Cdim * 2);    // transposed bf16
  __bf16* wproj   = (__bf16*)alloc((size_t)Cdim * Cdim * 2);
  __bf16* wfc1    = (__bf16*)alloc((size_t)Cdim * 4 * Cdim * 2);
  __bf16* wfc2    = (__bf16*)alloc((size_t)4 * Cdim * Cdim * 2);
  float*  pooled  = (float*) alloc((size_t)Bsz * Cdim * 4);

  // residual stream <- x
  hipMemcpyAsync(xres, d_in[0], (size_t)MROWS * Cdim * 4,
                 hipMemcpyDeviceToDevice, stream);

  for (int bi = 0; bi < 2; ++bi) {
    int base = 1 + bi * 13;
    const float* ln1_g  = (const float*)d_in[base + 0];
    const float* ln1_b  = (const float*)d_in[base + 1];
    const float* qkv_w  = (const float*)d_in[base + 2];
    const float* qkv_b  = (const float*)d_in[base + 3];
    const float* rpb    = (const float*)d_in[base + 4];
    const float* proj_w = (const float*)d_in[base + 5];
    const float* proj_b = (const float*)d_in[base + 6];
    const float* ln2_g  = (const float*)d_in[base + 7];
    const float* ln2_b  = (const float*)d_in[base + 8];
    const float* fc1_w  = (const float*)d_in[base + 9];
    const float* fc1_b  = (const float*)d_in[base + 10];
    const float* fc2_w  = (const float*)d_in[base + 11];
    const float* fc2_b  = (const float*)d_in[base + 12];
    int do_shift = bi;  // block 1 is shifted

    // convert weights to transposed bf16 [N,K]
    int nqkv = Cdim * 3 * Cdim, nproj = Cdim * Cdim, nfc = Cdim * 4 * Cdim;
    cvt_t_bf16_kernel<<<(nqkv + 255) / 256, 256, 0, stream>>>(qkv_w, wqkv, Cdim, 3 * Cdim);
    cvt_t_bf16_kernel<<<(nproj + 255) / 256, 256, 0, stream>>>(proj_w, wproj, Cdim, Cdim);
    cvt_t_bf16_kernel<<<(nfc + 255) / 256, 256, 0, stream>>>(fc1_w, wfc1, Cdim, 4 * Cdim);
    cvt_t_bf16_kernel<<<(nfc + 255) / 256, 256, 0, stream>>>(fc2_w, wfc2, 4 * Cdim, Cdim);

    // LN1 + (shifted) window partition
    ln_part_kernel<<<MROWS, 256, 0, stream>>>(xres, ln1_g, ln1_b, hbf, do_shift);
    // QKV GEMM (bf16 out)
    launch_gemm(hbf, wqkv, qkv_b, qkvbuf, MROWS, 3 * Cdim, Cdim, 1, 0, stream);
    // windowed attention
    attn_kernel<<<NWIN * NHEAD, 64, 0, stream>>>(qkvbuf, rpb, attnout, do_shift);
    // proj GEMM (f32 out)
    launch_gemm(attnout, wproj, proj_b, tmpf, MROWS, Cdim, Cdim, 0, 0, stream);
    // window reverse (+roll) + residual
    rev_res_kernel<<<MROWS, 256, 0, stream>>>(xres, tmpf, do_shift);
    // LN2
    ln_kernel<<<MROWS, 256, 0, stream>>>(xres, ln2_g, ln2_b, hbf);
    // MLP: fc1 (gelu, bf16 out) then fc2 (f32 out) + residual
    launch_gemm(hbf, wfc1, fc1_b, mlpbuf, MROWS, 4 * Cdim, Cdim, 1, 1, stream);
    launch_gemm(mlpbuf, wfc2, fc2_b, tmpf, MROWS, Cdim, 4 * Cdim, 0, 0, stream);
    int ntot = MROWS * Cdim;
    res_add_kernel<<<(ntot + 255) / 256, 256, 0, stream>>>(xres, tmpf, ntot);
  }

  // pool + head
  pool_kernel<<<Bsz, 256, 0, stream>>>(xres, pooled);
  head_kernel<<<Bsz, 256, 0, stream>>>(pooled,
                                       (const float*)d_in[27], (const float*)d_in[28],
                                       (const float*)d_in[29], (const float*)d_in[30],
                                       (const float*)d_in[31], (const float*)d_in[32],
                                       (const float*)d_in[33], (const float*)d_in[34],
                                       (float*)d_out);
}
